// GATNN_attpool_auto_14654428414344
// MI455X (gfx1250) — matrix operand order repaired and checked by reading, not compiled, not run on hardware
//
#include <hip/hip_runtime.h>
#include <hip/hip_bf16.h>

#define NNODES 30000
#define NEDGES 300000
#define NGRAPH 64
#define NHEAD  4
#define NCH    64
#define HC     256

typedef _Float16 v16h __attribute__((ext_vector_type(16)));
typedef _Float16 v8h  __attribute__((ext_vector_type(8)));
typedef float    v8f  __attribute__((ext_vector_type(8)));

__device__ __forceinline__ unsigned fenc(float f) {
  unsigned u = __float_as_uint(f);
  return (u & 0x80000000u) ? ~u : (u | 0x80000000u);
}
__device__ __forceinline__ float fdec(unsigned u) {
  unsigned b = (u & 0x80000000u) ? (u & 0x7fffffffu) : ~u;
  return __uint_as_float(b);
}
__device__ __forceinline__ float lrelu(float x, float s) { return x > 0.f ? x : s * x; }

// native fp32 global atomic add (global_atomic_add_f32) — avoid CAS fallback
__device__ __forceinline__ void atomAddF(float* p, float v) {
#if defined(__HIP_DEVICE_COMPILE__)
  unsafeAtomicAdd(p, v);
#else
  atomicAdd(p, v);
#endif
}

// ---------------- init node MLP: h0h = f16(x @ W0 + b0), [N,128] ----------------
__global__ void k_init_mlp(const float* __restrict__ x, const float* __restrict__ W0,
                           const float* __restrict__ b0, _Float16* __restrict__ h0h) {
  int idx = blockIdx.x * 256 + threadIdx.x;
  if (idx >= NNODES * 128) return;
  int n = idx >> 7, o = idx & 127;
  float acc = b0[o];
#pragma unroll
  for (int k = 0; k < 13; ++k) acc += x[n * 13 + k] * W0[k * 128 + o];
  h0h[idx] = (_Float16)acc;
}

// ---------------- pad + convert edge_attr: [E,51] f32 -> [E,64] f16 ----------------
__global__ void k_pad_ea(const float* __restrict__ ea, _Float16* __restrict__ eah) {
  int idx = blockIdx.x * 256 + threadIdx.x;
  if (idx >= NEDGES * 64) return;
  int e = idx >> 6, k = idx & 63;
  eah[idx] = (k < 51) ? (_Float16)ea[e * 51 + k] : (_Float16)0.f;
}

// ---------------- pack weight [Korig,Nc] f32 into WMMA B-fragment layout ----------------
// layout: idx = ((kt*Nc + n)*2 + half)*16 + e  maps K = kt*32 + (e&8)*2 + half*8 + (e&7)
__global__ void k_pack_w(const float* __restrict__ W, _Float16* __restrict__ Wp,
                         int Korig, int Kpad, int Nc) {
  int idx = blockIdx.x * 256 + threadIdx.x;
  if (idx >= Kpad * Nc) return;
  int e = idx & 15;
  int half = (idx >> 4) & 1;
  int rem = idx >> 5;
  int n = rem % Nc;
  int kt = rem / Nc;
  int k = kt * 32 + (e & 8) * 2 + half * 8 + (e & 7);
  float v = (k < Korig) ? W[k * Nc + n] : 0.f;
  Wp[idx] = (_Float16)v;
}

// ---------------- WMMA GEMM: C[M,Nc] = A_f16[M,K] @ Bpacked + bias ----------------
// grid (Nc/64, M/16), block 32: one wave computes a 16x64 strip (4 accumulators),
// so each A fragment is loaded once and reused for 4 WMMAs per k-step.
__global__ void k_wmma_gemm(const _Float16* __restrict__ A, const _Float16* __restrict__ Bp,
                            const float* __restrict__ bias, float* __restrict__ C,
                            int K, int Nc) {
  int lane = threadIdx.x;
  int half = lane >> 4;
  int l16 = lane & 15;
  int row0 = blockIdx.y * 16;
  int n0 = blockIdx.x * 64;
  const _Float16* Arow = A + (size_t)(row0 + l16) * K;
  v8f acc0 = {}, acc1 = {}, acc2 = {}, acc3 = {};
  for (int kt = 0; kt < (K >> 5); ++kt) {
    int kk = kt << 5;
    v8h alo = *(const v8h*)(Arow + kk + half * 8);
    v8h ahi = *(const v8h*)(Arow + kk + 16 + half * 8);
    v16h a = __builtin_shufflevector(alo, ahi, 0,1,2,3,4,5,6,7,8,9,10,11,12,13,14,15);
    const _Float16* bpk = Bp + (((size_t)kt * Nc + n0 + l16) * 2 + half) * 16;
    v16h b0 = *(const v16h*)(bpk + 0 * 32);   // n = n0 + 0*16 + l16
    v16h b1 = *(const v16h*)(bpk + 16 * 32);  // n = n0 + 1*16 + l16
    v16h b2 = *(const v16h*)(bpk + 32 * 32);  // n = n0 + 2*16 + l16
    v16h b3 = *(const v16h*)(bpk + 48 * 32);  // n = n0 + 3*16 + l16
    acc0 = __builtin_amdgcn_wmma_f32_16x16x32_f16(false, a, false, b0, (short)0, acc0, false, false);
    acc1 = __builtin_amdgcn_wmma_f32_16x16x32_f16(false, a, false, b1, (short)0, acc1, false, false);
    acc2 = __builtin_amdgcn_wmma_f32_16x16x32_f16(false, a, false, b2, (short)0, acc2, false, false);
    acc3 = __builtin_amdgcn_wmma_f32_16x16x32_f16(false, a, false, b3, (short)0, acc3, false, false);
  }
#pragma unroll
  for (int j = 0; j < 4; ++j) {
    v8f acc = (j == 0) ? acc0 : (j == 1) ? acc1 : (j == 2) ? acc2 : acc3;
    int n = n0 + j * 16 + l16;
    float bv = bias[n];
#pragma unroll
    for (int r = 0; r < 8; ++r)
      C[(size_t)(row0 + r + half * 8) * Nc + n] = acc[r] + bv;
  }
}

// ---------------- fills ----------------
__global__ void k_fill_u32(unsigned* __restrict__ p, unsigned v, int count) {
  int idx = blockIdx.x * 256 + threadIdx.x;
  if (idx < count) p[idx] = v;
}
__global__ void k_init_bias(float* __restrict__ out, const float* __restrict__ bias) {
  int idx = blockIdx.x * 256 + threadIdx.x;
  if (idx >= NNODES * HC) return;
  out[idx] = bias[idx & (HC - 1)];
}

// ---------------- edge logit kernel: ee via WMMA in LDS, then logits + seg-max ----------------
// block = 256 threads (8 waves), handles 16 edges. grid = E/16.
__global__ void k_edge_logit(const _Float16* __restrict__ eah, const _Float16* __restrict__ Wep,
                             const float* __restrict__ xl, const float* __restrict__ xr,
                             const int* __restrict__ src, const int* __restrict__ dst,
                             const float* __restrict__ att, float* __restrict__ logit,
                             unsigned* __restrict__ nmaxU) {
  __shared__ float ee[16 * HC];
  __shared__ float lg[64];
  int e0 = blockIdx.x * 16;
  int t = threadIdx.x;
  int wave = t >> 5;
  int lane = t & 31;
  int half = lane >> 4;
  int l16 = lane & 15;

  // ---- phase 1: ee tile [16 edges x 256 cols] via WMMA (K=64 -> 2 ksteps) ----
  const _Float16* Arow = eah + (size_t)(e0 + l16) * 64;
  v8h alo0 = *(const v8h*)(Arow + half * 8);
  v8h ahi0 = *(const v8h*)(Arow + 16 + half * 8);
  v16h a0 = __builtin_shufflevector(alo0, ahi0, 0,1,2,3,4,5,6,7,8,9,10,11,12,13,14,15);
  v8h alo1 = *(const v8h*)(Arow + 32 + half * 8);
  v8h ahi1 = *(const v8h*)(Arow + 48 + half * 8);
  v16h a1 = __builtin_shufflevector(alo1, ahi1, 0,1,2,3,4,5,6,7,8,9,10,11,12,13,14,15);
#pragma unroll
  for (int s = 0; s < 2; ++s) {
    int n = (wave * 2 + s) * 16 + l16;
    v16h b0 = *(const v16h*)(Wep + (((size_t)0 * HC + n) * 2 + half) * 16);
    v16h b1 = *(const v16h*)(Wep + (((size_t)1 * HC + n) * 2 + half) * 16);
    v8f acc = {};
    acc = __builtin_amdgcn_wmma_f32_16x16x32_f16(false, a0, false, b0, (short)0, acc, false, false);
    acc = __builtin_amdgcn_wmma_f32_16x16x32_f16(false, a1, false, b1, (short)0, acc, false, false);
#pragma unroll
    for (int r = 0; r < 8; ++r) ee[(r + half * 8) * HC + n] = acc[r];
  }
  if (t < 64) lg[t] = 0.f;
  __syncthreads();

  // ---- phase 2: z = lrelu(xl[src]+xr[dst]+ee), partial logit over 16 channels ----
  int e = t >> 4;
  int h = (t >> 2) & 3;
  int q = t & 3;
  int si = src[e0 + e];
  int di = dst[e0 + e];
  int cbase = h * 64 + q * 16;
  float partial = 0.f;
#pragma unroll
  for (int i = 0; i < 16; ++i) {
    int c = cbase + i;
    float z = lrelu(xl[(size_t)si * HC + c] + xr[(size_t)di * HC + c] + ee[e * HC + c], 0.2f);
    partial += z * att[h * 64 + q * 16 + i];
  }
  atomicAdd(&lg[e * 4 + h], partial);  // LDS: ds_add_f32
  __syncthreads();

  // ---- phase 3: write logit, segment-max via encoded atomicMax ----
  if (t < 64) {
    int ee2 = t >> 2, hh = t & 3;
    int d2 = dst[e0 + ee2];
    float v = lg[t];
    logit[(size_t)(e0 + ee2) * 4 + hh] = v;
    atomicMax(&nmaxU[(size_t)d2 * 4 + hh], fenc(v));
  }
}

// ---------------- w = exp(logit - m[dst]); den += w; logit := w ----------------
__global__ void k_edge_den(float* __restrict__ logit, const int* __restrict__ dst,
                           const unsigned* __restrict__ nmaxU, float* __restrict__ nden) {
  int idx = blockIdx.x * 256 + threadIdx.x;
  if (idx >= NEDGES * 4) return;
  int e = idx >> 2, h = idx & 3;
  int d = dst[e];
  float m = fdec(nmaxU[(size_t)d * 4 + h]);
  float w = __expf(logit[idx] - m);
  logit[idx] = w;
  atomAddF(&nden[(size_t)d * 4 + h], w);
}

// ---------------- scatter: out[dst] += xl[src] * alpha ---------------- (1 wave / edge)
__global__ void k_edge_scatter(const float* __restrict__ logit, const float* __restrict__ nden,
                               const float* __restrict__ xl, const int* __restrict__ src,
                               const int* __restrict__ dst, float* __restrict__ out) {
  int e = blockIdx.x * 8 + (threadIdx.x >> 5);
  int lane = threadIdx.x & 31;
  int si = src[e];
  int di = dst[e];
  float al[4];
#pragma unroll
  for (int h = 0; h < 4; ++h)
    al[h] = logit[(size_t)e * 4 + h] / (nden[(size_t)di * 4 + h] + 1e-16f);
#pragma unroll
  for (int j = 0; j < 8; ++j) {
    int c = lane + 32 * j;
    atomAddF(&out[(size_t)di * HC + c], xl[(size_t)si * HC + c] * al[j >> 1]);
  }
}

// ---------------- f32 -> f16 convert ----------------
__global__ void k_f32_to_f16(const float* __restrict__ in, _Float16* __restrict__ out, int count) {
  int idx = blockIdx.x * 256 + threadIdx.x;
  if (idx < count) out[idx] = (_Float16)in[idx];
}

// ---------------- pooling softmax over nodes per graph ----------------
__global__ void k_pool_max(const float* __restrict__ a, const int* __restrict__ batch,
                           unsigned* __restrict__ bmaxU) {
  int idx = blockIdx.x * 256 + threadIdx.x;
  if (idx >= NNODES * HC) return;
  int n = idx >> 8, c = idx & 255;
  atomicMax(&bmaxU[(size_t)batch[n] * HC + c], fenc(a[idx]));
}
__global__ void k_pool_den(float* __restrict__ a, const int* __restrict__ batch,
                           const unsigned* __restrict__ bmaxU, float* __restrict__ bden) {
  int idx = blockIdx.x * 256 + threadIdx.x;
  if (idx >= NNODES * HC) return;
  int n = idx >> 8, c = idx & 255;
  float m = fdec(bmaxU[(size_t)batch[n] * HC + c]);
  float w = __expf(a[idx] - m);
  a[idx] = w;
  atomAddF(&bden[(size_t)batch[n] * HC + c], w);
}
__global__ void k_pool_acc(const float* __restrict__ a, const float* __restrict__ h2,
                           const int* __restrict__ batch, const float* __restrict__ bden,
                           float* __restrict__ g) {
  int idx = blockIdx.x * 256 + threadIdx.x;
  if (idx >= NNODES * HC) return;
  int n = idx >> 8, c = idx & 255;
  int b = batch[n];
  float attw = a[idx] / (bden[(size_t)b * HC + c] + 1e-16f);
  atomAddF(&g[(size_t)b * HC + c], h2[idx] * attw);
}

// ---------------- small VALU GEMM with optional leaky-relu (slope=1 -> identity) ----------------
__global__ void k_small_gemm(const float* __restrict__ A, const float* __restrict__ W,
                             const float* __restrict__ bias, float* __restrict__ C,
                             int K, int Nc, int total, float slope) {
  int idx = blockIdx.x * 256 + threadIdx.x;
  if (idx >= total) return;
  int m = idx / Nc, o = idx - m * Nc;
  float acc = bias[o];
  for (int k = 0; k < K; ++k) acc += A[(size_t)m * K + k] * W[(size_t)k * Nc + o];
  C[idx] = lrelu(acc, slope);
}

// ---------------- batchnorm (train stats over 64 rows) + ReLU, single block ----------------
__global__ void k_bn_relu(float* __restrict__ y, const float* __restrict__ gamma,
                          const float* __restrict__ beta) {
  int c = threadIdx.x;  // 256 threads
  float mu = 0.f;
  for (int b = 0; b < NGRAPH; ++b) mu += y[b * HC + c];
  mu *= (1.f / NGRAPH);
  float var = 0.f;
  for (int b = 0; b < NGRAPH; ++b) {
    float d = y[b * HC + c] - mu;
    var += d * d;
  }
  var *= (1.f / NGRAPH);
  float inv = rsqrtf(var + 1e-5f);
  for (int b = 0; b < NGRAPH; ++b) {
    float v = (y[b * HC + c] - mu) * inv * gamma[c] + beta[c];
    y[b * HC + c] = v > 0.f ? v : 0.f;
  }
}

extern "C" void kernel_launch(void* const* d_in, const int* in_sizes, int n_in,
                              void* d_out, int out_size, void* d_ws, size_t ws_size,
                              hipStream_t stream) {
  const float* x   = (const float*)d_in[0];
  const int*   ei  = (const int*)d_in[1];
  const float* ea  = (const float*)d_in[2];
  const int*   bat = (const int*)d_in[3];
  const float* W0  = (const float*)d_in[4];
  const float* b0  = (const float*)d_in[5];
  const float* Wl1 = (const float*)d_in[6];
  const float* bl1 = (const float*)d_in[7];
  const float* Wr1 = (const float*)d_in[8];
  const float* br1 = (const float*)d_in[9];
  const float* We1 = (const float*)d_in[10];
  const float* att1 = (const float*)d_in[11];
  const float* bias1 = (const float*)d_in[12];
  const float* Wl2 = (const float*)d_in[13];
  const float* bl2 = (const float*)d_in[14];
  const float* Wr2 = (const float*)d_in[15];
  const float* br2 = (const float*)d_in[16];
  const float* We2 = (const float*)d_in[17];
  const float* att2 = (const float*)d_in[18];
  const float* bias2 = (const float*)d_in[19];
  const float* Wp  = (const float*)d_in[20];
  const float* bp  = (const float*)d_in[21];
  const float* Wo1 = (const float*)d_in[22];
  const float* bo1 = (const float*)d_in[23];
  const float* gam = (const float*)d_in[24];
  const float* bet = (const float*)d_in[25];
  const float* Wo2 = (const float*)d_in[26];
  const float* bo2 = (const float*)d_in[27];
  const int* src = ei;
  const int* dst = ei + NEDGES;

  // ---- workspace carve ----
  size_t off = 0;
  char* base = (char*)d_ws;
  auto carve = [&](size_t bytes) -> char* {
    char* p = base + off;
    off = (off + bytes + 255) & ~(size_t)255;
    return p;
  };
  _Float16* eah  = (_Float16*)carve((size_t)NEDGES * 64 * 2);
  _Float16* h0h  = (_Float16*)carve((size_t)NNODES * 128 * 2);
  float*    h1   = (float*)carve((size_t)NNODES * HC * 4);
  _Float16* h1h  = (_Float16*)carve((size_t)NNODES * HC * 2);
  float*    h2   = (float*)carve((size_t)NNODES * HC * 4);
  _Float16* h2h  = (_Float16*)carve((size_t)NNODES * HC * 2);
  float*    xl   = (float*)carve((size_t)NNODES * HC * 4);
  float*    xr   = (float*)carve((size_t)NNODES * HC * 4);
  float*    logit = (float*)carve((size_t)NEDGES * 4 * 4);
  unsigned* nmaxU = (unsigned*)carve((size_t)NNODES * 4 * 4);
  float*    nden  = (float*)carve((size_t)NNODES * 4 * 4);
  _Float16* Wl1p = (_Float16*)carve(128 * HC * 2);
  _Float16* Wr1p = (_Float16*)carve(128 * HC * 2);
  _Float16* We1p = (_Float16*)carve(64 * HC * 2);
  _Float16* Wl2p = (_Float16*)carve(HC * HC * 2);
  _Float16* Wr2p = (_Float16*)carve(HC * HC * 2);
  _Float16* We2p = (_Float16*)carve(64 * HC * 2);
  _Float16* Wpp  = (_Float16*)carve(HC * HC * 2);
  unsigned* bmaxU = (unsigned*)carve((size_t)NGRAPH * HC * 4);
  float*    bden  = (float*)carve((size_t)NGRAPH * HC * 4);
  float*    g     = (float*)carve((size_t)NGRAPH * HC * 4);
  float*    y     = (float*)carve((size_t)NGRAPH * HC * 4);
  if (off > ws_size) return;  // workspace too small; nothing safe to do

  auto blocks = [](long long n) { return (unsigned)((n + 255) / 256); };

  // ---- weight packing (f32 -> f16 WMMA B-fragment layout) ----
  k_pack_w<<<blocks(128 * HC), 256, 0, stream>>>(Wl1, Wl1p, 128, 128, HC);
  k_pack_w<<<blocks(128 * HC), 256, 0, stream>>>(Wr1, Wr1p, 128, 128, HC);
  k_pack_w<<<blocks(64 * HC), 256, 0, stream>>>(We1, We1p, 51, 64, HC);
  k_pack_w<<<blocks(HC * HC), 256, 0, stream>>>(Wl2, Wl2p, HC, HC, HC);
  k_pack_w<<<blocks(HC * HC), 256, 0, stream>>>(Wr2, Wr2p, HC, HC, HC);
  k_pack_w<<<blocks(64 * HC), 256, 0, stream>>>(We2, We2p, 51, 64, HC);
  k_pack_w<<<blocks(HC * HC), 256, 0, stream>>>(Wp, Wpp, HC, HC, HC);

  // ---- input prep ----
  k_init_mlp<<<blocks((long long)NNODES * 128), 256, 0, stream>>>(x, W0, b0, h0h);
  k_pad_ea<<<blocks((long long)NEDGES * 64), 256, 0, stream>>>(ea, eah);

  // ---- per-layer pipeline ----
  auto run_layer = [&](const _Float16* hin_h, int K, const _Float16* Wlp, const float* bl,
                       const _Float16* Wrp, const float* br, const _Float16* Wep,
                       const float* att, const float* bias, float* hout, _Float16* hout_h) {
    dim3 ggrid(HC / 64, NNODES / 16);
    k_wmma_gemm<<<ggrid, 32, 0, stream>>>(hin_h, Wlp, bl, xl, K, HC);
    k_wmma_gemm<<<ggrid, 32, 0, stream>>>(hin_h, Wrp, br, xr, K, HC);
    k_init_bias<<<blocks((long long)NNODES * HC), 256, 0, stream>>>(hout, bias);
    k_fill_u32<<<blocks((long long)NNODES * 4), 256, 0, stream>>>(nmaxU, 0u, NNODES * 4);
    k_fill_u32<<<blocks((long long)NNODES * 4), 256, 0, stream>>>((unsigned*)nden, 0u, NNODES * 4);
    k_edge_logit<<<NEDGES / 16, 256, 0, stream>>>(eah, Wep, xl, xr, src, dst, att, logit, nmaxU);
    k_edge_den<<<blocks((long long)NEDGES * 4), 256, 0, stream>>>(logit, dst, nmaxU, nden);
    k_edge_scatter<<<NEDGES / 8, 256, 0, stream>>>(logit, nden, xl, src, dst, hout);
    k_f32_to_f16<<<blocks((long long)NNODES * HC), 256, 0, stream>>>(hout, hout_h, NNODES * HC);
  };

  run_layer(h0h, 128, Wl1p, bl1, Wr1p, br1, We1p, att1, bias1, h1, h1h);
  run_layer(h1h, HC, Wl2p, bl2, Wr2p, br2, We2p, att2, bias2, h2, h2h);

  // ---- attention pooling over nodes per graph ----
  {
    dim3 ggrid(HC / 64, NNODES / 16);
    k_wmma_gemm<<<ggrid, 32, 0, stream>>>(h2h, Wpp, bp, xl, HC, HC);  // xl reused as scores 'a'
  }
  k_fill_u32<<<blocks((long long)NGRAPH * HC), 256, 0, stream>>>(bmaxU, 0u, NGRAPH * HC);
  k_fill_u32<<<blocks((long long)NGRAPH * HC), 256, 0, stream>>>((unsigned*)bden, 0u, NGRAPH * HC);
  k_fill_u32<<<blocks((long long)NGRAPH * HC), 256, 0, stream>>>((unsigned*)g, 0u, NGRAPH * HC);
  k_pool_max<<<blocks((long long)NNODES * HC), 256, 0, stream>>>(xl, bat, bmaxU);
  k_pool_den<<<blocks((long long)NNODES * HC), 256, 0, stream>>>(xl, bat, bmaxU, bden);
  k_pool_acc<<<blocks((long long)NNODES * HC), 256, 0, stream>>>(xl, h2, bat, bden, g);

  // ---- output MLP: Linear + BN(train) + ReLU + Linear + leaky_relu(0.01) ----
  k_small_gemm<<<blocks((long long)NGRAPH * HC), 256, 0, stream>>>(g, Wo1, bo1, y, HC, HC,
                                                                  NGRAPH * HC, 1.0f);
  k_bn_relu<<<1, 256, 0, stream>>>(y, gam, bet);
  k_small_gemm<<<blocks((long long)NGRAPH * 2001), 256, 0, stream>>>(y, Wo2, bo2, (float*)d_out,
                                                                    HC, 2001, NGRAPH * 2001, 0.01f);
}
